// SResnetNM_35167192220213
// MI455X (gfx1250) — matrix operand order repaired and checked by reading, not compiled, not run on hardware
//
#include <hip/hip_runtime.h>
#include <cstdint>
#include <cstddef>

// ---------------------------------------------------------------------------
// SResnet (spiking ResNet, n=2, F=64, T=6, B=16) forward on MI455X (gfx1250).
// Conv layers run as implicit GEMM in f16 via v_wmma_f32_16x16x32_f16 with
// 16Mx64N register blocking per wave (4 accumulators, A-fragment reuse).
// Activations (spikes + skip sums) are exactly representable in f16.
// All tensors use channel-major [C, B, H, W] layout so WMMA epilogue stores
// and BatchNorm reductions are contiguous per channel.
// ---------------------------------------------------------------------------

#define B_    16
#define T_    6
#define LEAK_ 0.95f
#define EPS_  1e-4f

typedef __attribute__((ext_vector_type(16))) _Float16 v16h;
typedef __attribute__((ext_vector_type(8)))  _Float16 v8h;
typedef __attribute__((ext_vector_type(8)))  float    v8f;
typedef __attribute__((ext_vector_type(4)))  float    v4f;

union V16H { v16h v; v8h h[2]; };
union V8F  { v8f  v; v4f q[2]; };

// ---- weight f32 -> f16, [Co][K] -> [Co][Kpad] zero-padded --------------------
__global__ void sres_wconvert(const float* __restrict__ w, _Float16* __restrict__ wf,
                              int K, int Kpad, int total) {
  int idx = blockIdx.x * 256 + threadIdx.x;
  if (idx >= total) return;
  int co = idx / Kpad, k = idx - co * Kpad;
  wf[idx] = (k < K) ? (_Float16)w[co * K + k] : (_Float16)0.0f;
}

// ---- input x[B,T,2,64,64] f32 -> act[2,B,64,64] f16 (one timestep) ----------
__global__ void sres_input_convert(const float* __restrict__ x, _Float16* __restrict__ act, int t) {
  const int HW = 64 * 64;
  int idx = blockIdx.x * 256 + threadIdx.x;          // 2*B_*HW
  if (idx >= 2 * B_ * HW) return;
  int c = idx / (B_ * HW); int rem = idx - c * (B_ * HW);
  int b = rem / HW;        int p = rem - b * HW;
  act[idx] = (_Float16)x[(((size_t)b * T_ + t) * 2 + c) * HW + p];
}

// ---- im2col: act[Ci,B,H,W] f16 -> col[M,Kpad] f16 ---------------------------
__global__ void sres_im2col(const _Float16* __restrict__ act, _Float16* __restrict__ col,
                            int Ci, int H, int W, int Ho, int Wo,
                            int stride, int ks, int pad, int K, int Kpad, long total) {
  long idx = (long)blockIdx.x * 256 + threadIdx.x;
  if (idx >= total) return;
  int  k = (int)(idx % Kpad);
  long m = idx / Kpad;
  _Float16 v = (_Float16)0.0f;
  if (k < K) {
    int kss = ks * ks;
    int ci = k / kss; int rs = k - ci * kss;
    int r = rs / ks, s = rs - r * ks;
    int hw = Ho * Wo;
    int b = (int)(m / hw); int p = (int)(m - (long)b * hw);
    int oy = p / Wo, ox = p - oy * Wo;
    int iy = oy * stride + r - pad, ix = ox * stride + s - pad;
    if ((unsigned)iy < (unsigned)H && (unsigned)ix < (unsigned)W)
      v = act[(((size_t)ci * B_ + b) * H + iy) * W + ix];
  }
  col[idx] = v;
}

// ---- WMMA GEMM: C[N,M] f32 = col[M,Kpad] * Bw[N,Kpad]^T ---------------------
// 8 waves / workgroup; each wave computes a 16(M) x 64(N) strip as four
// 16x16 WMMA tiles, reusing one A fragment per K-step (4 WMMAs / 10 b128
// loads). grid = (M/128, N/64) so the col matrix is re-read only N/64 times.
// Fragment layouts used here (contiguous k order, same for A and B):
//   A 16x32 f16: lanes 0-15 row m hold K kk..kk+15;
//                lanes 16-31 hold K kk+16..kk+31.
//   B 32x16 f16: lane n (0-15) = column n, K kk..kk+15; lane 16+n = K kk+16..kk+31.
//   C/D f32: lane n = column n; acc[r] = row r (lanes<16) / 8+r (lanes>=16).
__global__ __launch_bounds__(256)
void sres_gemm_wmma(const _Float16* __restrict__ A, const _Float16* __restrict__ Bw,
                    float* __restrict__ C, int M, int Kpad) {
  const int lane  = threadIdx.x & 31;
  const int wave  = threadIdx.x >> 5;
  const int mbase = blockIdx.x * 128 + wave * 16;
  const int nbase = blockIdx.y * 64;
  const int l15   = lane & 15;
  const int hi    = lane >> 4;

  const _Float16* ap = A  + (size_t)(mbase + l15) * Kpad + (hi << 4);
  const _Float16* bp = Bw + (size_t)(nbase + l15) * Kpad + (hi << 4);
  const size_t bs = (size_t)16 * Kpad;               // stride between N tiles

  v8f acc0 = {}, acc1 = {}, acc2 = {}, acc3 = {};
  for (int kk = 0; kk < Kpad; kk += 32) {
    V16H a, b0, b1, b2, b3;
    a.h[0]  = *(const v8h*)(ap);
    a.h[1]  = *(const v8h*)(ap + 8);
    b0.h[0] = *(const v8h*)(bp);
    b0.h[1] = *(const v8h*)(bp + 8);
    b1.h[0] = *(const v8h*)(bp + bs);
    b1.h[1] = *(const v8h*)(bp + bs + 8);
    b2.h[0] = *(const v8h*)(bp + 2 * bs);
    b2.h[1] = *(const v8h*)(bp + 2 * bs + 8);
    b3.h[0] = *(const v8h*)(bp + 3 * bs);
    b3.h[1] = *(const v8h*)(bp + 3 * bs + 8);
    ap += 32; bp += 32;
    __builtin_prefetch(ap, 0, 1);                    // global_prefetch_b8
    __builtin_prefetch(bp, 0, 1);
    acc0 = __builtin_amdgcn_wmma_f32_16x16x32_f16(false, a.v, false, b0.v,
                                                  (short)0, acc0, false, false);
    acc1 = __builtin_amdgcn_wmma_f32_16x16x32_f16(false, a.v, false, b1.v,
                                                  (short)0, acc1, false, false);
    acc2 = __builtin_amdgcn_wmma_f32_16x16x32_f16(false, a.v, false, b2.v,
                                                  (short)0, acc2, false, false);
    acc3 = __builtin_amdgcn_wmma_f32_16x16x32_f16(false, a.v, false, b3.v,
                                                  (short)0, acc3, false, false);
  }
  const size_t cb = (size_t)(nbase + l15) * M + mbase + (hi << 3);
  V8F o;
  o.v = acc0; { float* cp = C + cb;              *(v4f*)cp = o.q[0]; *(v4f*)(cp + 4) = o.q[1]; }
  o.v = acc1; { float* cp = C + cb + 16 * (size_t)M; *(v4f*)cp = o.q[0]; *(v4f*)(cp + 4) = o.q[1]; }
  o.v = acc2; { float* cp = C + cb + 32 * (size_t)M; *(v4f*)cp = o.q[0]; *(v4f*)(cp + 4) = o.q[1]; }
  o.v = acc3; { float* cp = C + cb + 48 * (size_t)M; *(v4f*)cp = o.q[0]; *(v4f*)(cp + 4) = o.q[1]; }
}

// ---- per-channel batch stats (mean, biased var) over M ----------------------
__global__ __launch_bounds__(256)
void sres_bn_stats(const float* __restrict__ conv, float* __restrict__ stats, int M, int C) {
  __shared__ float sh[512];
  int c = blockIdx.x;
  const float* p = conv + (size_t)c * M;
  float s = 0.f, s2 = 0.f;
  for (int i = threadIdx.x; i < M; i += 256) { float v = p[i]; s += v; s2 += v * v; }
  sh[threadIdx.x] = s; sh[256 + threadIdx.x] = s2;
  __syncthreads();
  for (int off = 128; off > 0; off >>= 1) {
    if ((int)threadIdx.x < off) {
      sh[threadIdx.x]       += sh[threadIdx.x + off];
      sh[256 + threadIdx.x] += sh[256 + threadIdx.x + off];
    }
    __syncthreads();
  }
  if (threadIdx.x == 0) {
    float mu  = sh[0]   / (float)M;
    float var = sh[256] / (float)M - mu * mu;
    stats[c] = mu; stats[C + c] = var;
  }
}

// ---- fused BN + leaky membrane + spike + soft reset + skip add --------------
__global__ void sres_update(const float* __restrict__ conv, const float* __restrict__ stats,
                            const float* __restrict__ gamma, const float* __restrict__ beta,
                            float* __restrict__ mem, const _Float16* __restrict__ skip,
                            _Float16* __restrict__ out, int C, int M) {
  long idx = (long)blockIdx.x * 256 + threadIdx.x;
  if (idx >= (long)C * M) return;
  int c = (int)(idx / M);
  float bnv  = (conv[idx] - stats[c]) * rsqrtf(stats[C + c] + EPS_) * gamma[c] + beta[c];
  float memv = LEAK_ * mem[idx] + bnv;
  float thr  = memv - 1.0f;                      // mem/THRESH - 1, THRESH=1
  float spk  = (thr > 0.f) ? 1.f : 0.f;
  mem[idx] = memv - spk;                         // soft reset by THRESH
  float o = spk + (skip ? (float)skip[idx] : 0.f);
  out[idx] = (_Float16)o;
}

// ---- BN apply only (skip-projection path) -----------------------------------
__global__ void sres_bn_apply(const float* __restrict__ conv, const float* __restrict__ stats,
                              const float* __restrict__ gamma, const float* __restrict__ beta,
                              _Float16* __restrict__ out, int C, int M) {
  long idx = (long)blockIdx.x * 256 + threadIdx.x;
  if (idx >= (long)C * M) return;
  int c = (int)(idx / M);
  float bnv = (conv[idx] - stats[c]) * rsqrtf(stats[C + c] + EPS_) * gamma[c] + beta[c];
  out[idx] = (_Float16)bnv;
}

// ---- spatial mean pool of out12[256,B,8,8], accumulate over timesteps -------
__global__ void sres_pool_acc(const _Float16* __restrict__ act, float* __restrict__ pooled, int HW) {
  int j = blockIdx.x * 256 + threadIdx.x;        // j = c*B + b, C=256
  if (j >= 256 * B_) return;
  const _Float16* p = act + (size_t)j * HW;
  float s = 0.f;
  for (int i = 0; i < HW; i++) s += (float)p[i];
  pooled[j] += s / (float)HW;
}

// ---- final FC: out[b,cls] = (1/T) * sum_c pooled[c,b] * fcw[cls,c] ----------
__global__ void sres_fc(const float* __restrict__ pooled, const float* __restrict__ fcw,
                        float* __restrict__ out) {
  int i = threadIdx.x;
  if (i >= B_ * 10) return;
  int b = i / 10, cls = i - b * 10;
  float s = 0.f;
  for (int c = 0; c < 256; c++) s += pooled[c * B_ + b] * fcw[cls * 256 + c];
  out[i] = s / (float)T_;
}

// ---------------------------------------------------------------------------
// Host orchestration.
// Input order (setup_inputs dict order, lists expanded in sequence):
//   d_in[0]      x                                 [16,6,2,64,64]
//   d_in[1..13]  conv_w[0..12]                     [Co,Ci,3,3]
//   d_in[14..26] bn_gamma[0..12]                   [6,Co]
//   d_in[27..39] bn_beta[0..12]                    [6,Co]
//   d_in[40..41] resize_w[0..1]                    [2F,F,1,1],[4F,2F,1,1]
//   d_in[42..43] resize_gamma[0..1]                [6,2F],[6,4F]
//   d_in[44..45] resize_beta[0..1]
//   d_in[46]     fc_w                              [10,256]
// ---------------------------------------------------------------------------
extern "C" void kernel_launch(void* const* d_in, const int* in_sizes, int n_in,
                              void* d_out, int out_size, void* d_ws, size_t ws_size,
                              hipStream_t stream) {
  (void)in_sizes; (void)n_in; (void)out_size; (void)ws_size;

  const float* x = (const float*)d_in[0];
  const float* conv_w[13]; const float* bn_g[13]; const float* bn_b[13];
  for (int i = 0; i < 13; i++) {
    conv_w[i] = (const float*)d_in[1 + i];
    bn_g[i]   = (const float*)d_in[14 + i];
    bn_b[i]   = (const float*)d_in[27 + i];
  }
  const float* rw[2]  = {(const float*)d_in[40], (const float*)d_in[41]};
  const float* rg[2]  = {(const float*)d_in[42], (const float*)d_in[43]};
  const float* rb[2]  = {(const float*)d_in[44], (const float*)d_in[45]};
  const float* fcw    = (const float*)d_in[46];

  // Layer geometry.
  static const int Ci[13] = {2, 64, 64, 64, 64, 64, 128, 128, 128, 128, 256, 256, 256};
  static const int Co[13] = {64, 64, 64, 64, 64, 128, 128, 128, 128, 256, 256, 256, 256};
  static const int Hi[13] = {64, 32, 32, 32, 32, 32, 16, 16, 16, 16, 8, 8, 8};
  static const int St[13] = {2, 1, 1, 1, 1, 2, 1, 1, 1, 2, 1, 1, 1};
  int Ho[13], K[13], Kp[13], M[13];
  for (int i = 0; i < 13; i++) {
    Ho[i] = Hi[i] / St[i];
    K[i]  = Ci[i] * 9;
    Kp[i] = (K[i] + 31) & ~31;
    M[i]  = B_ * Ho[i] * Ho[i];       // 16384 / 4096 / 1024 — all %128 == 0
  }
  // Resize (skip projection) convs: 1x1 stride 2.
  static const int rCi[2] = {64, 128}, rCo[2] = {128, 256};
  static const int rHi[2] = {32, 16},  rHo[2] = {16, 8};
  int rM[2]; for (int r = 0; r < 2; r++) rM[r] = B_ * rHo[r] * rHo[r];

  // Workspace bump allocator (total ~83 MB).
  char* ws = (char*)d_ws;
  size_t off = 0;
  auto alloc = [&](size_t bytes) -> size_t {
    size_t o = off; off += (bytes + 255) & ~(size_t)255; return o;
  };

  size_t wfO[13], memO[13], outO[13];
  for (int i = 0; i < 13; i++) wfO[i] = alloc((size_t)Co[i] * Kp[i] * 2);
  size_t rwO[2]; for (int r = 0; r < 2; r++) rwO[r] = alloc((size_t)rCo[r] * rCi[r] * 2);

  size_t colMax = 0, convMax = 0;
  for (int i = 0; i < 13; i++) {
    size_t ce = (size_t)M[i] * Kp[i]; if (ce > colMax) colMax = ce;
    size_t cv = (size_t)Co[i] * M[i]; if (cv > convMax) convMax = cv;
  }
  size_t colO   = alloc(colMax * 2);
  size_t convO  = alloc(convMax * 4);
  size_t conv2O = alloc((size_t)rCo[0] * rM[0] * 4);   // biggest projection conv
  size_t statsO  = alloc(2 * 256 * 4);
  size_t stats2O = alloc(2 * 256 * 4);
  for (int i = 0; i < 13; i++) memO[i] = alloc((size_t)Co[i] * M[i] * 4);
  size_t actInO = alloc((size_t)2 * B_ * 64 * 64 * 2);
  for (int i = 0; i < 13; i++) outO[i] = alloc((size_t)Co[i] * M[i] * 2);
  size_t projO[2]; for (int r = 0; r < 2; r++) projO[r] = alloc((size_t)rCo[r] * rM[r] * 2);
  size_t pooledO = alloc((size_t)256 * B_ * 4);

  _Float16* col   = (_Float16*)(ws + colO);
  float*    convb = (float*)(ws + convO);
  float*    convr = (float*)(ws + conv2O);
  float*    stats  = (float*)(ws + statsO);
  float*    stats2 = (float*)(ws + stats2O);
  _Float16* actIn  = (_Float16*)(ws + actInO);
  float*    pooled = (float*)(ws + pooledO);

  // One-time (per launch, deterministic): weight conversion + state zeroing.
  for (int i = 0; i < 13; i++) {
    int tot = Co[i] * Kp[i];
    sres_wconvert<<<(tot + 255) / 256, 256, 0, stream>>>(conv_w[i], (_Float16*)(ws + wfO[i]),
                                                         K[i], Kp[i], tot);
  }
  for (int r = 0; r < 2; r++) {
    int tot = rCo[r] * rCi[r];
    sres_wconvert<<<(tot + 255) / 256, 256, 0, stream>>>(rw[r], (_Float16*)(ws + rwO[r]),
                                                         rCi[r], rCi[r], tot);
  }
  for (int i = 0; i < 13; i++)
    hipMemsetAsync(ws + memO[i], 0, (size_t)Co[i] * M[i] * 4, stream);
  hipMemsetAsync(ws + pooledO, 0, (size_t)256 * B_ * 4, stream);

  // Timestep loop (jax.lax.scan equivalent).
  for (int t = 0; t < T_; t++) {
    sres_input_convert<<<(2 * B_ * 4096 + 255) / 256, 256, 0, stream>>>(x, actIn, t);

    const _Float16* prev = actIn;
    const _Float16* skip = nullptr;

    for (int i = 0; i < 13; i++) {
      // conv_i as implicit GEMM
      long ctot = (long)M[i] * Kp[i];
      sres_im2col<<<(unsigned)((ctot + 255) / 256), 256, 0, stream>>>(
          prev, col, Ci[i], Hi[i], Hi[i], Ho[i], Ho[i], St[i], 3, 1, K[i], Kp[i], ctot);
      dim3 g((unsigned)(M[i] / 128), (unsigned)(Co[i] / 64));
      sres_gemm_wmma<<<g, 256, 0, stream>>>(col, (const _Float16*)(ws + wfO[i]),
                                            convb, M[i], Kp[i]);
      sres_bn_stats<<<Co[i], 256, 0, stream>>>(convb, stats, M[i], Co[i]);

      // Skip handling (even layers >= 2 add skip; 6/10 run projection first).
      const _Float16* skipArg = nullptr;
      if (i > 0 && (i & 1) == 0) {
        if (i == 6 || i == 10) {
          int r = (i == 6) ? 0 : 1;
          long rtot = (long)rM[r] * rCi[r];                 // Kpad == Ci (1x1)
          sres_im2col<<<(unsigned)((rtot + 255) / 256), 256, 0, stream>>>(
              skip, col, rCi[r], rHi[r], rHi[r], rHo[r], rHo[r], 2, 1, 0,
              rCi[r], rCi[r], rtot);
          dim3 gr((unsigned)(rM[r] / 128), (unsigned)(rCo[r] / 64));
          sres_gemm_wmma<<<gr, 256, 0, stream>>>(col, (const _Float16*)(ws + rwO[r]),
                                                 convr, rM[r], rCi[r]);
          sres_bn_stats<<<rCo[r], 256, 0, stream>>>(convr, stats2, rM[r], rCo[r]);
          long ptot = (long)rCo[r] * rM[r];
          sres_bn_apply<<<(unsigned)((ptot + 255) / 256), 256, 0, stream>>>(
              convr, stats2, rg[r] + t * rCo[r], rb[r] + t * rCo[r],
              (_Float16*)(ws + projO[r]), rCo[r], rM[r]);
          skipArg = (const _Float16*)(ws + projO[r]);
        } else {
          skipArg = skip;
        }
      }

      long utot = (long)Co[i] * M[i];
      sres_update<<<(unsigned)((utot + 255) / 256), 256, 0, stream>>>(
          convb, stats, bn_g[i] + t * Co[i], bn_b[i] + t * Co[i],
          (float*)(ws + memO[i]), skipArg, (_Float16*)(ws + outO[i]), Co[i], M[i]);

      if ((i & 1) == 0) skip = (const _Float16*)(ws + outO[i]);
      prev = (const _Float16*)(ws + outO[i]);
    }

    sres_pool_acc<<<(256 * B_ + 255) / 256, 256, 0, stream>>>(
        (const _Float16*)(ws + outO[12]), pooled, 8 * 8);
  }

  sres_fc<<<1, 256, 0, stream>>>(pooled, fcw, (float*)d_out);
}
